// TokenTopKMoeBlock_69664369541777
// MI455X (gfx1250) — matrix operand order repaired and checked by reading, not compiled, not run on hardware
//
#include <hip/hip_runtime.h>
#include <hip/hip_bf16.h>
#include <math.h>

#define T_TOKENS 4096
#define H_DIM    1024
#define F_DIM    4096
#define N_EXP    8
#define TILE_M   64                  // rows per tile (4 WMMA M-tiles)
#define XPITCH   (H_DIM + 8)         // bf16 elems; +16B breaks LDS bank conflicts
#define NROWS    (2 * T_TOKENS)      // total gathered rows (each token in 2 lists)
#define NROWS_PAD (NROWS + TILE_M)   // slack for tail over-read

typedef __attribute__((ext_vector_type(16))) __bf16 v16bf;
typedef __attribute__((ext_vector_type(8)))  __bf16 bf16x8;
typedef __attribute__((ext_vector_type(8)))  float  v8f;

// ---- CDNA5 async global->LDS copy (ASYNCcnt) ----
__device__ __forceinline__ void async_copy_b128(const void* gptr, void* lptr) {
  unsigned lo = (unsigned)(size_t)lptr;  // LDS aperture: low 32 bits = LDS offset
  asm volatile("global_load_async_to_lds_b128 %0, %1, off"
               :: "v"(lo), "v"(gptr) : "memory");
}
__device__ __forceinline__ void async_wait0() {
  asm volatile("s_wait_asynccnt 0x0" ::: "memory");
}

// A-matrix (16xK bf16) fragment for one 16x16x32 step.
// Lane l: row m = l&15, half = l>>4.
// elements 0..7: k = k0 + half*16 + i ; elements 8..15: k = k0 + half*16 + 8 + i
__device__ __forceinline__ v16bf load_a_frag(const __bf16* row, int k0, int half) {
  bf16x8 lo = *(const bf16x8*)(row + k0 + half * 16);
  bf16x8 hi = *(const bf16x8*)(row + k0 + half * 16 + 8);
  v16bf a;
#pragma unroll
  for (int i = 0; i < 8; ++i) { a[i] = lo[i]; a[i + 8] = hi[i]; }
  return a;
}

// B-matrix (Kx16 bf16) fragment: lane l = column n (l&15); 16 consecutive k
// values at (k0 + half*16) -> one contiguous 32B load per lane.
__device__ __forceinline__ v16bf load_b_frag_bf16(const __bf16* col, int k) {
  return *(const v16bf*)(col + k);
}

__global__ __launch_bounds__(256) void zero_kernel(float* out, int* counts) {
  size_t stride = (size_t)gridDim.x * blockDim.x;
  size_t i = (size_t)blockIdx.x * blockDim.x + threadIdx.x;
  float4 z = {0.f, 0.f, 0.f, 0.f};
  float4* o = (float4*)out;
  for (size_t j = i; j < (size_t)T_TOKENS * H_DIM / 4; j += stride) o[j] = z;
  if (blockIdx.x == 0 && threadIdx.x < 16) counts[threadIdx.x] = 0;
}

// One-shot fp32 -> bf16 weight conversion into workspace (memory bound).
__global__ __launch_bounds__(256) void cvt_kernel(const float* __restrict__ src,
                                                  __bf16* __restrict__ dst, size_t n) {
  size_t i = ((size_t)blockIdx.x * blockDim.x + threadIdx.x) * 8;
  size_t stride = (size_t)gridDim.x * blockDim.x * 8;
  for (; i < n; i += stride) {
    float4 a = *(const float4*)(src + i);
    float4 b = *(const float4*)(src + i + 4);
    bf16x8 o;
    o[0] = (__bf16)a.x; o[1] = (__bf16)a.y; o[2] = (__bf16)a.z; o[3] = (__bf16)a.w;
    o[4] = (__bf16)b.x; o[5] = (__bf16)b.y; o[6] = (__bf16)b.z; o[7] = (__bf16)b.w;
    *(bf16x8*)(dst + i) = o;
  }
}

__global__ __launch_bounds__(256) void router_kernel(
    const float* __restrict__ x, const float* __restrict__ wr,
    float* __restrict__ logits_out, int* __restrict__ counts,
    int* __restrict__ idx_list, float* __restrict__ gate_list) {
  const int t = blockIdx.x;
  const int w = threadIdx.x >> 5, lane = threadIdx.x & 31;
  __shared__ float lg[N_EXP];
  const float* xt = x + (size_t)t * H_DIM;
  const float* we = wr + (size_t)w * H_DIM;
  float s = 0.f;
  for (int i = lane; i < H_DIM; i += 32) s += xt[i] * we[i];
#pragma unroll
  for (int off = 16; off > 0; off >>= 1) s += __shfl_xor(s, off);
  if (lane == 0) { lg[w] = s; logits_out[(size_t)t * N_EXP + w] = s; }
  __syncthreads();
  if (threadIdx.x == 0) {
    float l[N_EXP];
#pragma unroll
    for (int e = 0; e < N_EXP; ++e) l[e] = lg[e];
    int e1 = 0;
#pragma unroll
    for (int e = 1; e < N_EXP; ++e) if (l[e] > l[e1]) e1 = e;
    int e2 = (e1 == 0) ? 1 : 0;
#pragma unroll
    for (int e = 0; e < N_EXP; ++e) if (e != e1 && l[e] > l[e2]) e2 = e;
    float p2 = __expf(l[e2] - l[e1]);
    float inv = 1.f / (1.f + p2);
    int p = atomicAdd(&counts[e1], 1);
    idx_list[e1 * T_TOKENS + p] = t; gate_list[e1 * T_TOKENS + p] = inv;
    p = atomicAdd(&counts[e2], 1);
    idx_list[e2 * T_TOKENS + p] = t; gate_list[e2 * T_TOKENS + p] = p2 * inv;
  }
}

__global__ void offsets_kernel(const int* __restrict__ counts, int* __restrict__ offs) {
  if (blockIdx.x == 0 && threadIdx.x == 0) {
    int s = 0;
#pragma unroll
    for (int e = 0; e < N_EXP; ++e) { offs[e] = s; s += counts[e]; }
    offs[N_EXP] = s;                 // == 2T
  }
}

// Compact gathered activations: Xg[row] = bf16(x[token(row)]), rows grouped by expert.
__global__ __launch_bounds__(128) void gather_kernel(
    const float* __restrict__ x, const int* __restrict__ offs,
    const int* __restrict__ idx_list, __bf16* __restrict__ Xg) {
  const int r = blockIdx.x;          // 0 .. 2T-1
  int e = 0;
  while (e < N_EXP - 1 && r >= offs[e + 1]) ++e;
  const int tok = idx_list[e * T_TOKENS + (r - offs[e])];
  const float* src = x + (size_t)tok * H_DIM;
  __bf16* dst = Xg + (size_t)r * H_DIM;
  for (int i = threadIdx.x * 8; i < H_DIM; i += 128 * 8) {
    float4 a = *(const float4*)(src + i);
    float4 b = *(const float4*)(src + i + 4);
    bf16x8 o;
    o[0] = (__bf16)a.x; o[1] = (__bf16)a.y; o[2] = (__bf16)a.z; o[3] = (__bf16)a.w;
    o[4] = (__bf16)b.x; o[5] = (__bf16)b.y; o[6] = (__bf16)b.z; o[7] = (__bf16)b.w;
    *(bf16x8*)(dst + i) = o;
  }
}

// Kernel A: H = silu(Xg@w1^T) * (Xg@w3^T), bf16 out. Block: 64 rows x 512 F-cols.
__global__ __launch_bounds__(512) void moe_h_kernel(
    const __bf16* __restrict__ Xg, const __bf16* __restrict__ wb1,
    const __bf16* __restrict__ wb3, __bf16* __restrict__ H,
    const int* __restrict__ counts, const int* __restrict__ offs) {
  const int e = blockIdx.z;
  const int cnt = counts[e];
  const int base = blockIdx.x * TILE_M;
  if (base >= cnt) return;
  const int cntLoc = (cnt - base < TILE_M) ? (cnt - base) : TILE_M;
  const int rowbase = offs[e] + base;

  __shared__ alignas(16) __bf16 xa[TILE_M * XPITCH];   // 129 KB
  const int tid = threadIdx.x;
  // async-stage 64 contiguous Xg rows into padded LDS
#pragma unroll
  for (int i = 0; i < 16; ++i) {
    int c = tid + 512 * i;                  // 8192 chunks of 8 bf16 (16B)
    int row = c >> 7, co = (c & 127) * 8;
    async_copy_b128(Xg + (size_t)(rowbase + row) * H_DIM + co,
                    &xa[row * XPITCH + co]);
  }
  async_wait0();
  __syncthreads();

  const int w = tid >> 5, lane = tid & 31;
  const int nl = lane & 15, half = lane >> 4;
  const int fbase = blockIdx.y * 512 + w * 32;   // this wave: 32 F-cols, both mats
  const __bf16* w1c0 = wb1 + (size_t)e * F_DIM * H_DIM + (size_t)(fbase + nl) * H_DIM;
  const __bf16* w3c0 = wb3 + (size_t)e * F_DIM * H_DIM + (size_t)(fbase + nl) * H_DIM;

  v8f zero = {};
  v8f acc1[4][2], acc3[4][2];
#pragma unroll
  for (int mt = 0; mt < 4; ++mt)
#pragma unroll
    for (int i = 0; i < 2; ++i) { acc1[mt][i] = zero; acc3[mt][i] = zero; }

#pragma unroll 2
  for (int k0 = 0; k0 < H_DIM; k0 += 32) {
    const int kk = k0 + half * 16;
    v16bf a[4];
#pragma unroll
    for (int mt = 0; mt < 4; ++mt)
      a[mt] = load_a_frag(&xa[(mt * 16 + nl) * XPITCH], k0, half);
    v16bf b10 = load_b_frag_bf16(w1c0, kk);
    v16bf b11 = load_b_frag_bf16(w1c0 + (size_t)16 * H_DIM, kk);
    v16bf b30 = load_b_frag_bf16(w3c0, kk);
    v16bf b31 = load_b_frag_bf16(w3c0 + (size_t)16 * H_DIM, kk);
#pragma unroll
    for (int mt = 0; mt < 4; ++mt) {
      acc1[mt][0] = __builtin_amdgcn_wmma_f32_16x16x32_bf16(false, a[mt], false, b10,
                                                            (short)0, acc1[mt][0], false, false);
      acc1[mt][1] = __builtin_amdgcn_wmma_f32_16x16x32_bf16(false, a[mt], false, b11,
                                                            (short)0, acc1[mt][1], false, false);
      acc3[mt][0] = __builtin_amdgcn_wmma_f32_16x16x32_bf16(false, a[mt], false, b30,
                                                            (short)0, acc3[mt][0], false, false);
      acc3[mt][1] = __builtin_amdgcn_wmma_f32_16x16x32_bf16(false, a[mt], false, b31,
                                                            (short)0, acc3[mt][1], false, false);
    }
  }

  // silu(h1)*h3 in registers -> bf16 H (guard rows beyond this expert's count)
#pragma unroll
  for (int mt = 0; mt < 4; ++mt)
#pragma unroll
    for (int i = 0; i < 2; ++i)
#pragma unroll
      for (int r = 0; r < 8; ++r) {
        int m = mt * 16 + r + half * 8;
        if (m < cntLoc) {
          float v1 = acc1[mt][i][r], v3 = acc3[mt][i][r];
          H[(size_t)(rowbase + m) * F_DIM + fbase + i * 16 + nl] =
              (__bf16)((v1 / (1.f + __expf(-v1))) * v3);
        }
      }
}

// Kernel B: out += gate * (H @ w2^T). Block: 64 rows x all 1024 cols.
// Double-buffered async LDS staging of contiguous H K-chunks.
__global__ __launch_bounds__(512) void moe_out_kernel(
    const __bf16* __restrict__ H, const __bf16* __restrict__ wb2,
    float* __restrict__ out, const int* __restrict__ counts,
    const int* __restrict__ offs, const int* __restrict__ idx_list,
    const float* __restrict__ gate_list) {
  const int e = blockIdx.y;
  const int cnt = counts[e];
  const int base = blockIdx.x * TILE_M;
  if (base >= cnt) return;
  const int cntLoc = (cnt - base < TILE_M) ? (cnt - base) : TILE_M;
  const int rowbase = offs[e] + base;

  __shared__ alignas(16) __bf16 ha0[TILE_M * XPITCH];  // 129 KB
  __shared__ alignas(16) __bf16 ha1[TILE_M * XPITCH];  // 129 KB
  __shared__ int   toks[TILE_M];
  __shared__ float gts[TILE_M];

  const int tid = threadIdx.x;
  if (tid < TILE_M) {
    if (tid < cntLoc) {
      toks[tid] = idx_list[e * T_TOKENS + base + tid];
      gts[tid]  = gate_list[e * T_TOKENS + base + tid];
    } else { toks[tid] = -1; gts[tid] = 0.f; }
  }

  const int w = tid >> 5, lane = tid & 31;
  const int nl = lane & 15, half = lane >> 4;
  const int nbase = w * 64;
  const __bf16* wb2e = wb2 + (size_t)e * H_DIM * F_DIM;

  // stage K-chunk kc (1024 of F) into buffer
  auto stage = [&](__bf16* buf, int kc) {
#pragma unroll
    for (int i = 0; i < 16; ++i) {
      int c = tid + 512 * i;
      int row = c >> 7, co = (c & 127) * 8;
      async_copy_b128(H + (size_t)(rowbase + row) * F_DIM + kc * H_DIM + co,
                      &buf[row * XPITCH + co]);
    }
  };

  v8f zero = {};
  v8f acc[4][4];
#pragma unroll
  for (int mt = 0; mt < 4; ++mt)
#pragma unroll
    for (int nt = 0; nt < 4; ++nt) acc[mt][nt] = zero;

  stage(ha0, 0);
  async_wait0();
  __syncthreads();

  for (int kc = 0; kc < F_DIM / H_DIM; ++kc) {
    if (kc + 1 < F_DIM / H_DIM) stage((kc & 1) ? ha0 : ha1, kc + 1);
    const __bf16* ha = (kc & 1) ? ha1 : ha0;
#pragma unroll 2
    for (int s = 0; s < H_DIM / 32; ++s) {
      const int kglob = kc * H_DIM + s * 32 + half * 16;
#pragma unroll
      for (int mt = 0; mt < 4; ++mt) {
        v16bf a = load_a_frag(&ha[(mt * 16 + nl) * XPITCH], s * 32, half);
#pragma unroll
        for (int nt = 0; nt < 4; ++nt) {
          const __bf16* colB = wb2e + (size_t)(nbase + nt * 16 + nl) * F_DIM;
          v16bf b = load_b_frag_bf16(colB, kglob);
          acc[mt][nt] = __builtin_amdgcn_wmma_f32_16x16x32_bf16(
              false, a, false, b, (short)0, acc[mt][nt], false, false);
        }
      }
    }
    if (kc + 1 < F_DIM / H_DIM) { async_wait0(); __syncthreads(); }
  }

  // gate-weighted scatter (exactly 2 addends per out element -> deterministic)
#pragma unroll
  for (int mt = 0; mt < 4; ++mt)
#pragma unroll
    for (int nt = 0; nt < 4; ++nt)
#pragma unroll
      for (int r = 0; r < 8; ++r) {
        int m = mt * 16 + r + half * 8;
        if (m < cntLoc) {
          atomicAdd(out + (size_t)toks[m] * H_DIM + nbase + nt * 16 + nl,
                    gts[m] * acc[mt][nt][r]);
        }
      }
}

extern "C" void kernel_launch(void* const* d_in, const int* in_sizes, int n_in,
                              void* d_out, int out_size, void* d_ws, size_t ws_size,
                              hipStream_t stream) {
  const float* x  = (const float*)d_in[0];
  const float* wr = (const float*)d_in[1];
  const float* w1 = (const float*)d_in[2];
  const float* w2 = (const float*)d_in[3];
  const float* w3 = (const float*)d_in[4];

  float* out    = (float*)d_out;                        // [T, H]
  float* logits = out + (size_t)T_TOKENS * H_DIM;       // [T, E]

  const size_t WELEM = (size_t)N_EXP * F_DIM * H_DIM;   // 33.5M elems per matrix
  int*    counts = (int*)d_ws;                          // 16 ints
  int*    offs   = counts + 16;                         // 16 ints (uses 9)
  int*    idx_l  = offs + 16;                           // [E, T]
  float*  gate_l = (float*)(idx_l + N_EXP * T_TOKENS);  // [E, T]
  __bf16* wb1    = (__bf16*)(gate_l + N_EXP * T_TOKENS);
  __bf16* wb3    = wb1 + WELEM;
  __bf16* wb2    = wb3 + WELEM;
  __bf16* Xg     = wb2 + WELEM;                         // [NROWS_PAD, H]
  __bf16* Hbuf   = Xg + (size_t)NROWS_PAD * H_DIM;      // [NROWS_PAD, F]

  zero_kernel<<<1024, 256, 0, stream>>>(out, counts);
  cvt_kernel<<<8192, 256, 0, stream>>>(w1, wb1, WELEM);
  cvt_kernel<<<8192, 256, 0, stream>>>(w3, wb3, WELEM);
  cvt_kernel<<<8192, 256, 0, stream>>>(w2, wb2, WELEM);
  router_kernel<<<T_TOKENS, 256, 0, stream>>>(x, wr, logits, counts, idx_l, gate_l);
  offsets_kernel<<<1, 32, 0, stream>>>(counts, offs);
  gather_kernel<<<NROWS, 128, 0, stream>>>(x, offs, idx_l, Xg);
  moe_h_kernel<<<dim3(T_TOKENS / TILE_M, F_DIM / 512, N_EXP), 512, 0, stream>>>(
      Xg, wb1, wb3, Hbuf, counts, offs);
  moe_out_kernel<<<dim3(T_TOKENS / TILE_M, N_EXP), 512, 0, stream>>>(
      Hbuf, wb2, out, counts, offs, idx_l, gate_l);
}
